// Critic_74045236183049
// MI455X (gfx1250) — matrix operand order repaired and checked by reading, not compiled, not run on hardware
//
#include <hip/hip_runtime.h>

typedef __attribute__((ext_vector_type(16))) _Float16 v16h;
typedef __attribute__((ext_vector_type(8)))  _Float16 v8h;
typedef __attribute__((ext_vector_type(8)))  float    v8f;

#define HH   256
#define NN   128
#define NEGC 9000000000000000.0f
#define SCALEC (1.0f / (64.0f * 36.0f))

// Explicit global address space so loads lower to global_load_* (not flat_load_*).
typedef __attribute__((address_space(1))) const float    gcf;
typedef __attribute__((address_space(1))) const v16h     gcv16h;
typedef __attribute__((address_space(1)))       float    gf;

static __device__ __forceinline__ float ldg_f(const float* p) {
  return *(gcf*)p;
}

static __device__ __forceinline__ v8f wmma_f16(v16h a, v16h b, v8f c) {
  // D = A(16x32 f16) * B(32x16 f16) + C(16x16 f32)
  return __builtin_amdgcn_wmma_f32_16x16x32_f16(
      /*neg_a=*/false, a, /*neg_b=*/false, b,
      /*c_mod=*/(short)0, c, /*reuse_a=*/false, /*reuse_b=*/false);
}

// A fragment (16x32, f16) from row-major f16 LDS buffer with row stride ld.
// Layout: lanes 0-15 row M=lane, elems 0..15 = K k0..k0+15;
//         lanes 16-31 row M=lane-16, K k0+16..k0+31.
static __device__ __forceinline__ v16h load_a_f16(const _Float16* base, int row0, int ld,
                                                  int k0, int lane) {
  int r  = row0 + (lane & 15);
  int kk = k0 + ((lane >> 4) << 4);
  const _Float16* p = base + r * ld + kk;
  v8h lo = *(const v8h*)p;
  v8h hi = *(const v8h*)(p + 8);
  v16h a;
#pragma unroll
  for (int i = 0; i < 8; ++i) { a[i] = lo[i]; a[i + 8] = hi[i]; }
  return a;
}

// B fragment (32x16, f16) from column-major LDS layout buf[n*ld + k].
// Layout (contiguous k, same as load_a_f16): lanes 0-15 = col N=lane, K k0..k0+15; lanes 16-31 = col N=lane-16, K k0+16..k0+31.
static __device__ __forceinline__ v16h load_b_lds(const _Float16* base, int col0, int ld,
                                                  int k0, int lane) {
  int n  = col0 + (lane & 15);
  int kk = k0 + ((lane >> 4) << 4);
  return *(const v16h*)(base + n * ld + kk);
}

// Same B fragment but from global memory (weights): force addrspace(1).
static __device__ __forceinline__ v16h load_b_g(const _Float16* base, int col0, int ld,
                                                int k0, int lane) {
  int n  = col0 + (lane & 15);
  int kk = k0 + ((lane >> 4) << 4);
  return *(gcv16h*)(base + n * ld + kk);
}

// One attention block, fully LDS-resident for a single batch row.
//  sh  : h in / h' out             (f16 [128][256])
//  sq  : q, then f32 scores        (f16 [128][256] == f32 [128][128])
//  sk  : k, then f16 att after softmax (f16 [128][256]; att uses first 32KB)
//  svT : v transposed              (f16 [256][128])
static __device__ __forceinline__ void att_block(
    _Float16* sh, _Float16* sq, _Float16* sk, _Float16* svT,
    const _Float16* WvT, const float* bv,
    const _Float16* WkT, const float* bk,
    const _Float16* WqT, const float* bq,
    const float* maskB, int wave, int lane, int tid) {
  // ---- 1) q,k,v = relu(h @ W + b).  Wave owns n-tiles {2w, 2w+1}; for each,
  //         two groups of 4 m-tiles so each weight B-frag is reused 4x. ----
  for (int nn = 0; nn < 2; ++nn) {
    const int nt = wave * 2 + nn;
    for (int mg = 0; mg < 2; ++mg) {
      v8f cq[4] = {}, ck[4] = {}, cv[4] = {};
      for (int k0 = 0; k0 < HH; k0 += 32) {
        v16h bQ = load_b_g(WqT, nt * 16, HH, k0, lane);
        v16h bK = load_b_g(WkT, nt * 16, HH, k0, lane);
        v16h bV = load_b_g(WvT, nt * 16, HH, k0, lane);
#pragma unroll
        for (int m = 0; m < 4; ++m) {
          v16h a = load_a_f16(sh, (mg * 4 + m) * 16, HH, k0, lane);
          cq[m] = wmma_f16(a, bQ, cq[m]);
          ck[m] = wmma_f16(a, bK, ck[m]);
          cv[m] = wmma_f16(a, bV, cv[m]);
        }
      }
      int n = nt * 16 + (lane & 15);
      float vbq = ldg_f(bq + n), vbk = ldg_f(bk + n), vbv = ldg_f(bv + n);
#pragma unroll
      for (int m = 0; m < 4; ++m) {
        int m0 = (mg * 4 + m) * 16 + ((lane >> 4) << 3);
        v8h vpack;
#pragma unroll
        for (int j = 0; j < 8; ++j) {
          sq[(m0 + j) * HH + n] = (_Float16)fmaxf(cq[m][j] + vbq, 0.0f);
          sk[(m0 + j) * HH + n] = (_Float16)fmaxf(ck[m][j] + vbk, 0.0f);
          vpack[j] = (_Float16)fmaxf(cv[m][j] + vbv, 0.0f);
        }
        *(v8h*)(svT + n * NN + m0) = vpack;  // transposed v store: contiguous 16B
      }
    }
  }
  __syncthreads();

  // ---- 2) scores = SCALE * (q @ k^T): wave owns one 16-row band, A reused 8x ----
  v8f sc[8] = {};
  for (int k0 = 0; k0 < HH; k0 += 32) {
    v16h a = load_a_f16(sq, wave * 16, HH, k0, lane);
#pragma unroll
    for (int nt = 0; nt < 8; ++nt) {
      v16h bK = load_b_lds(sk, nt * 16, HH, k0, lane);  // k row-major == B for k^T
      sc[nt] = wmma_f16(a, bK, sc[nt]);
    }
  }
  __syncthreads();  // all q reads done -> sq reusable as f32 scores
  float* scF = (float*)sq;
  {
    int m0 = wave * 16 + ((lane >> 4) << 3);
#pragma unroll
    for (int nt = 0; nt < 8; ++nt) {
      int n = nt * 16 + (lane & 15);
#pragma unroll
      for (int j = 0; j < 8; ++j) scF[(m0 + j) * NN + n] = sc[nt][j] * SCALEC;
    }
  }
  __syncthreads();

  // ---- 3) masked softmax: 2 threads per row, combined with lane-pair shuffles;
  //         result written as f16 attention matrix into sk's (dead) buffer ----
  _Float16* satt = sk;  // f16 [128][128]
  {
    int row  = tid >> 1;
    int half = (tid & 1) << 6;  // 0 or 64
    const float* mrow = maskB + (size_t)row * NN + half;
    float* srow = scF + row * NN + half;
    float mx = -__builtin_inff();
    for (int j = 0; j < 64; ++j) {
      float m = ldg_f(mrow + j);
      float s = srow[j] * m - NEGC * (1.0f - m);
      srow[j] = s;
      mx = fmaxf(mx, s);
    }
    mx = fmaxf(mx, __shfl_xor(mx, 1));
    float sum = 0.0f;
    for (int j = 0; j < 64; ++j) {
      float e = __expf(srow[j] - mx);
      srow[j] = e;
      sum += e;
    }
    sum += __shfl_xor(sum, 1);
    float inv = 1.0f / sum;
    _Float16* arow = satt + row * NN + half;
    for (int j = 0; j < 64; ++j) arow[j] = (_Float16)(srow[j] * inv);
  }
  __syncthreads();

  // ---- 4) h' = relu(att @ v): wave owns n-tiles {2w,2w+1}; each v^T B-frag
  //         reused across all 8 m-tiles ----
  for (int nn = 0; nn < 2; ++nn) {
    const int nt = wave * 2 + nn;
    v8f c[8] = {};
    for (int k0 = 0; k0 < NN; k0 += 32) {
      v16h bV = load_b_lds(svT, nt * 16, NN, k0, lane);
#pragma unroll
      for (int m = 0; m < 8; ++m) {
        v16h a = load_a_f16(satt, m * 16, NN, k0, lane);
        c[m] = wmma_f16(a, bV, c[m]);
      }
    }
    int n = nt * 16 + (lane & 15);
#pragma unroll
    for (int m = 0; m < 8; ++m) {
      int m0 = m * 16 + ((lane >> 4) << 3);
#pragma unroll
      for (int j = 0; j < 8; ++j) sh[(m0 + j) * HH + n] = (_Float16)fmaxf(c[m][j], 0.0f);
    }
  }
  __syncthreads();
}

// Convert f32 row-major W[K][N] -> f16 column-major Wt[n*K + k].
__global__ void prep_transpose(const float* __restrict__ w, _Float16* __restrict__ wt,
                               int K, int N) {
  int i = blockIdx.x * blockDim.x + threadIdx.x;
  if (i < K * N) {
    int k = i / N, n = i % N;
    wt[n * K + k] = (_Float16)(*(gcf*)(w + i));
  }
}

__global__ __launch_bounds__(256) void critic_kernel(
    const float* __restrict__ ob, const float* __restrict__ ac,
    const float* __restrict__ mask, const float* __restrict__ We,
    const float* __restrict__ be, const float* __restrict__ bv1,
    const float* __restrict__ bk1, const float* __restrict__ bq1,
    const float* __restrict__ bv2, const float* __restrict__ bk2,
    const float* __restrict__ bq2, const float* __restrict__ bf1,
    const float* __restrict__ Wf, const float* __restrict__ bf,
    const _Float16* __restrict__ wt, float* __restrict__ out) {
  extern __shared__ __align__(32) _Float16 smem[];
  _Float16* sh  = smem;                 // h   f16 [128][256] 64KB
  _Float16* sq  = smem + 32768;         // q / f32 scores     64KB
  _Float16* sk  = smem + 65536;         // k / f16 att / f32 hout 64KB
  _Float16* svT = smem + 98304;         // v^T f16 [256][128] 64KB
  __shared__ float sx[NN * 10];

  const int tid  = threadIdx.x;
  const int wave = tid >> 5;
  const int lane = tid & 31;
  const int b    = blockIdx.x;
  const size_t rowBase = (size_t)b * NN;

  // ---- stage x = concat(ob[:, :6], ac) into LDS ----
  for (int i = tid; i < NN * 6; i += 256) {
    int r = i / 6, c = i % 6;
    sx[r * 10 + c] = ldg_f(ob + (rowBase + r) * 6 + c);
  }
  for (int i = tid; i < NN * 4; i += 256) {
    int r = i / 4, c = i % 4;
    sx[r * 10 + 6 + c] = ldg_f(ac + (rowBase + r) * 4 + c);
  }
  __syncthreads();

  // ---- encoder: h1 = relu(x @ We + be); one output column per thread ----
  {
    float wcol[10];
#pragma unroll
    for (int f = 0; f < 10; ++f) wcol[f] = ldg_f(We + f * HH + tid);
    float bcol = ldg_f(be + tid);
    for (int r = 0; r < NN; ++r) {
      float acc = bcol;
#pragma unroll
      for (int f = 0; f < 10; ++f) acc += sx[r * 10 + f] * wcol[f];
      sh[r * HH + tid] = (_Float16)fmaxf(acc, 0.0f);
    }
  }
  __syncthreads();

  const float* maskB = mask + (size_t)b * NN * NN;

  att_block(sh, sq, sk, svT, wt + 0 * 65536, bv1, wt + 1 * 65536, bk1,
            wt + 2 * 65536, bq1, maskB, wave, lane, tid);
  att_block(sh, sq, sk, svT, wt + 3 * 65536, bv2, wt + 4 * 65536, bk2,
            wt + 5 * 65536, bq2, maskB, wave, lane, tid);

  // ---- head: hout = relu(h @ Wf1 + bf1) via WMMA (8x4 tiles, 4/wave) ----
  const _Float16* Wf1T = wt + 6 * 65536;  // f16 [64][256]
  float* houtF = (float*)sk;              // f32 [128][64]
  for (int i = 0; i < 4; ++i) {
    int t = wave * 4 + i;
    int nt = t & 3, mt = t >> 2;
    v8f c = {};
    for (int k0 = 0; k0 < HH; k0 += 32) {
      v16h a  = load_a_f16(sh, mt * 16, HH, k0, lane);
      v16h bm = load_b_g(Wf1T, nt * 16, HH, k0, lane);
      c = wmma_f16(a, bm, c);
    }
    int n  = nt * 16 + (lane & 15);
    int m0 = mt * 16 + ((lane >> 4) << 3);
    float bb = ldg_f(bf1 + n);
#pragma unroll
    for (int j = 0; j < 8; ++j) houtF[(m0 + j) * 64 + n] = fmaxf(c[j] + bb, 0.0f);
  }
  __syncthreads();

  // ---- final: out = hout @ Wf + bf; 2 threads per row + lane-pair shuffle ----
  {
    int row = tid >> 1;
    int j0  = (tid & 1) << 5;  // 0 or 32
    float acc = 0.0f;
#pragma unroll 8
    for (int j = 0; j < 32; ++j) acc += houtF[row * 64 + j0 + j] * ldg_f(Wf + j0 + j);
    acc += __shfl_xor(acc, 1);
    if ((tid & 1) == 0) *(gf*)(out + rowBase + row) = acc + ldg_f(bf);
  }
}

extern "C" void kernel_launch(void* const* d_in, const int* in_sizes, int n_in,
                              void* d_out, int out_size, void* d_ws, size_t ws_size,
                              hipStream_t stream) {
  const float* ob   = (const float*)d_in[0];
  const float* ac   = (const float*)d_in[1];
  const float* mask = (const float*)d_in[2];
  const float* We   = (const float*)d_in[3];
  const float* be   = (const float*)d_in[4];
  const float* Wv1  = (const float*)d_in[5];
  const float* bv1  = (const float*)d_in[6];
  const float* Wk1  = (const float*)d_in[7];
  const float* bk1  = (const float*)d_in[8];
  const float* Wq1  = (const float*)d_in[9];
  const float* bq1  = (const float*)d_in[10];
  const float* Wv2  = (const float*)d_in[11];
  const float* bv2  = (const float*)d_in[12];
  const float* Wk2  = (const float*)d_in[13];
  const float* bk2  = (const float*)d_in[14];
  const float* Wq2  = (const float*)d_in[15];
  const float* bq2  = (const float*)d_in[16];
  const float* Wf1  = (const float*)d_in[17];
  const float* bf1  = (const float*)d_in[18];
  const float* Wf   = (const float*)d_in[19];
  const float* bf   = (const float*)d_in[20];
  float* out = (float*)d_out;

  _Float16* wt = (_Float16*)d_ws;  // 6x[256*256] + [64*256] halves = 800KB

  prep_transpose<<<256, 256, 0, stream>>>(Wv1, wt + 0 * 65536, 256, 256);
  prep_transpose<<<256, 256, 0, stream>>>(Wk1, wt + 1 * 65536, 256, 256);
  prep_transpose<<<256, 256, 0, stream>>>(Wq1, wt + 2 * 65536, 256, 256);
  prep_transpose<<<256, 256, 0, stream>>>(Wv2, wt + 3 * 65536, 256, 256);
  prep_transpose<<<256, 256, 0, stream>>>(Wk2, wt + 4 * 65536, 256, 256);
  prep_transpose<<<256, 256, 0, stream>>>(Wq2, wt + 5 * 65536, 256, 256);
  prep_transpose<<<64, 256, 0, stream>>>(Wf1, wt + 6 * 65536, 256, 64);

  critic_kernel<<<1024, 256, 262144, stream>>>(ob, ac, mask, We, be, bv1, bk1, bq1,
                                               bv2, bk2, bq2, bf1, Wf, bf, wt, out);
}
